// SequenceFeaturesAttention_66984309948698
// MI455X (gfx1250) — matrix operand order repaired and checked by reading, not compiled, not run on hardware
//
#include <hip/hip_runtime.h>
#include <hip/hip_bf16.h>
#include <cstdint>
#include <cstddef>

#define S_DIM 128
#define L_DIM 512
#define D_DIM 256
#define H_DIM 8
#define C_DIM 32
#define HC_DIM 256
#define SL_DIM (S_DIM * L_DIM)

typedef __attribute__((ext_vector_type(16))) _Float16     v16h;
typedef __attribute__((ext_vector_type(8)))  _Float16     v8h;
typedef __attribute__((ext_vector_type(2)))  _Float16     h2;
typedef __attribute__((ext_vector_type(8)))  float        v8f;
typedef __attribute__((ext_vector_type(4)))  unsigned int u32x4;
typedef __attribute__((ext_vector_type(8)))  unsigned int u32x8;

// ---------------------------------------------------------------------------
// WMMA fragment helpers (CDNA5 16x16x32 f16, wave32)
// ---------------------------------------------------------------------------
__device__ __forceinline__ v16h load_a_frag(const _Float16* base, int ld) {
  const int lane = threadIdx.x & 31;
  const int m = lane & 15, g = lane >> 4;
  v16h a;
#pragma unroll
  for (int r = 0; r < 8; ++r) {
    const int kk = (r < 4) ? (2 * r + 8 * g) : (16 + 2 * (r - 4) + 8 * g);
    h2 t = *(const h2*)(base + m * ld + kk);
    a[2 * r] = t[0];
    a[2 * r + 1] = t[1];
  }
  return a;
}

__device__ __forceinline__ v16h load_bT_frag(const _Float16* baseT, int ld) {
  const int lane = threadIdx.x & 31;
  const int n = lane & 15, g = lane >> 4;
  v16h b;
#pragma unroll
  for (int r = 0; r < 8; ++r) {
    const int kk = (r < 4) ? (2 * r + 8 * g) : (16 + 2 * (r - 4) + 8 * g);
    h2 t = *(const h2*)(baseT + n * ld + kk);
    b[2 * r] = t[0];
    b[2 * r + 1] = t[1];
  }
  return b;
}

__device__ __forceinline__ v8f wmma_f16(v16h a, v16h b, v8f c) {
  return __builtin_amdgcn_wmma_f32_16x16x32_f16(false, a, false, b,
                                                (short)0, c, false, false);
}

// ---------------------------------------------------------------------------
// CDNA5 async / tensor data-movement primitives
// ---------------------------------------------------------------------------
// 16B-per-lane async copy global -> LDS (ASYNCcnt tracked).
__device__ __forceinline__ void async_copy_b128(unsigned int lds_byte_off,
                                                unsigned int voff_bytes,
                                                const void* sbase) {
  asm volatile("global_load_async_to_lds_b128 %0, %1, %2"
               ::"v"(lds_byte_off), "v"(voff_bytes), "s"(sbase)
               : "memory");
}
__device__ __forceinline__ void wait_async0() {
  asm volatile("s_wait_asynccnt 0x0" ::: "memory");
}

// TDM: 2D tile (tile_d0 elems wide x tile_d1 rows, f16) from global
// (row stride `stride_elems` halves) into LDS. Descriptor per ISA 8.3/8.4.
__device__ __forceinline__ void tdm_load_2d_f16(unsigned int lds_byte_off,
                                                const _Float16* gptr,
                                                unsigned int tile_d0,
                                                unsigned int tile_d1,
                                                unsigned int stride_elems) {
  const unsigned long long ga = (unsigned long long)(const void*)gptr;
  u32x4 g0;
  g0[0] = 1u;                                   // count=1, user mode
  g0[1] = lds_byte_off;                         // lds_addr
  g0[2] = (unsigned int)ga;                     // global_addr[31:0]
  g0[3] = ((unsigned int)(ga >> 32) & 0x01FFFFFFu) | (2u << 30);  // type=2
  u32x8 g1;
  g1[0] = 1u << 16;                             // data_size = 1 (2 bytes)
  g1[1] = tile_d0 << 16;                        // tensor_dim0[15:0]
  g1[2] = tile_d1 << 16;                        // tensor_dim1[15:0]
  g1[3] = tile_d0 << 16;                        // tile_dim0
  g1[4] = tile_d1;                              // tile_dim1 (tile_dim2=0)
  g1[5] = stride_elems;                         // tensor_dim0_stride[31:0]
  g1[6] = 0u;
  g1[7] = 0u;
  asm volatile("tensor_load_to_lds %0, %1" ::"s"(g0), "s"(g1) : "memory");
}

// ---------------------------------------------------------------------------
// Kernel 1: LayerNorm (one wave per row of 256), output f16
// ---------------------------------------------------------------------------
__global__ void ln_kernel(const float* __restrict__ x_in,
                          const float* __restrict__ ln_scale,
                          const float* __restrict__ ln_bias,
                          _Float16* __restrict__ x16) {
  const int wave = threadIdx.x >> 5;
  const int lane = threadIdx.x & 31;
  const int row = blockIdx.x * 8 + wave;
  const float* xr = x_in + (size_t)row * D_DIM;

  float v[8];
  float s = 0.f, s2 = 0.f;
#pragma unroll
  for (int i = 0; i < 8; ++i) {
    v[i] = xr[lane + 32 * i];
    s += v[i];
    s2 += v[i] * v[i];
  }
#pragma unroll
  for (int mask = 16; mask >= 1; mask >>= 1) {
    s += __shfl_xor(s, mask, 32);
    s2 += __shfl_xor(s2, mask, 32);
  }
  const float mu = s * (1.0f / D_DIM);
  const float var = s2 * (1.0f / D_DIM) - mu * mu;
  const float inv = rsqrtf(var + 1e-5f);
#pragma unroll
  for (int i = 0; i < 8; ++i) {
    const int c = lane + 32 * i;
    x16[(size_t)row * D_DIM + c] =
        (_Float16)((v[i] - mu) * inv * ln_scale[c] + ln_bias[c]);
  }
}

// ---------------------------------------------------------------------------
// Kernel 2: pack wq|wk|wv|wg into fused f16 [256,1024], wo into f16 [256,256]
// ---------------------------------------------------------------------------
__global__ void pack_w_kernel(const float* __restrict__ wq,
                              const float* __restrict__ wk,
                              const float* __restrict__ wv,
                              const float* __restrict__ wg,
                              const float* __restrict__ wo,
                              _Float16* __restrict__ Wf,
                              _Float16* __restrict__ wo16) {
  const int idx = blockIdx.x * 256 + threadIdx.x;
  if (idx < 262144) {
    const int d = idx >> 10, c = idx & 1023;
    float v;
    if (c < 256)      v = wq[d * 256 + c];
    else if (c < 512) v = wk[d * 256 + (c - 256)];
    else if (c < 768) v = wv[d * 256 + (c - 512)];
    else              v = wg[d * 256 + (c - 768)];
    Wf[idx] = (_Float16)v;
  } else if (idx < 262144 + 65536) {
    const int j = idx - 262144;
    wo16[j] = (_Float16)wo[j];
  }
}

// ---------------------------------------------------------------------------
// Kernels 3/5: tiled WMMA GEMM, C[M,N] = A[M,256] * B[256,N]
// 64x128 block tile, 8 waves; wave -> 16x64 strip = 4 WMMA tiles per k-step.
// Double-buffered LDS; A staged via async global->LDS copies; fully unrolled
// K loop (8 steps x 4 WMMAs). F32OUT: add bias, write f32; else f16.
// ---------------------------------------------------------------------------
template <int N, bool F32OUT>
__global__ void gemm_kernel(const _Float16* __restrict__ A,
                            const _Float16* __restrict__ B,
                            _Float16* __restrict__ Ch,
                            float* __restrict__ Cf,
                            const float* __restrict__ bias) {
  __shared__ __align__(16) _Float16 As[2][64 * 32];    // row-major [m][k]
  __shared__ __align__(16) _Float16 Bs[2][128 * 32];   // transposed [n][k]

  const int tid = threadIdx.x;
  const int lane = tid & 31, wave = tid >> 5;
  const int wr = wave & 3, wc = wave >> 2;   // 4 row strips x 2 col halves
  const int m0 = blockIdx.y * 64;
  const int n0 = blockIdx.x * 128;

  // staging coordinates
  const int ar = tid >> 2;             // A row 0..63
  const int ac = (tid & 3) * 8;        // A col group (halves)
  const int bk = (tid & 15) * 2;       // B k-pair 0,2,..,30
  const int bn = (tid >> 4) * 8;       // B n group 0..120
  const unsigned int asBase = (unsigned int)(size_t)(void*)&As[0][0];

  auto stage = [&](int kb, int buf) {
    // A tile: async 16B/lane copy straight into LDS
    const unsigned int lds =
        asBase + (unsigned int)(buf * 64 * 32 * 2 + (ar * 32 + ac) * 2);
    const unsigned int voff =
        (unsigned int)((((size_t)(m0 + ar)) * 256 + kb + ac) * 2);
    async_copy_b128(lds, voff, (const void*)A);
    // B tile: 2 k-rows x 8 n-cols per thread, store transposed as h2 packs
    v8h t0 = *(const v8h*)(B + (size_t)(kb + bk) * N + n0 + bn);
    v8h t1 = *(const v8h*)(B + (size_t)(kb + bk + 1) * N + n0 + bn);
#pragma unroll
    for (int i = 0; i < 8; ++i) {
      h2 p;
      p[0] = t0[i];
      p[1] = t1[i];
      *(h2*)(&Bs[buf][(bn + i) * 32 + bk]) = p;
    }
  };

  v8f acc[4] = {{}, {}, {}, {}};

  stage(0, 0);
  wait_async0();
  __syncthreads();

#pragma unroll
  for (int i = 0; i < 8; ++i) {
    if (i + 1 < 8) stage((i + 1) * 32, (i + 1) & 1);

    const _Float16* as = &As[i & 1][0];
    const _Float16* bs = &Bs[i & 1][0];
    v16h af = load_a_frag(as + wr * 16 * 32, 32);
#pragma unroll
    for (int j = 0; j < 4; ++j) {
      v16h bf = load_bT_frag(bs + (wc * 64 + j * 16) * 32, 32);
      acc[j] = wmma_f16(af, bf, acc[j]);
    }

    wait_async0();
    __syncthreads();
  }

  const int nn = lane & 15, g = lane >> 4;
#pragma unroll
  for (int j = 0; j < 4; ++j) {
#pragma unroll
    for (int r = 0; r < 8; ++r) {
      const int row = m0 + wr * 16 + r + 8 * g;
      const int col = n0 + wc * 64 + j * 16 + nn;
      if (F32OUT) {
        Cf[(size_t)row * N + col] = acc[j][r] + bias[col];
      } else {
        Ch[(size_t)row * N + col] = (_Float16)acc[j][r];
      }
    }
  }
}

// ---------------------------------------------------------------------------
// Kernel 4: attention over MSA-row axis, one block per (head, residue).
// Q/K/G tiles staged via Tensor Data Mover; V staged transposed via VGPRs.
// ---------------------------------------------------------------------------
__global__ void attn_kernel(const _Float16* __restrict__ qkvg,
                            const float* __restrict__ bg,
                            _Float16* __restrict__ obuf) {
  const int h = blockIdx.x;
  const int l = blockIdx.y;

  __shared__ __align__(16) _Float16 Qs[128 * 32];   // [s][c]
  __shared__ __align__(16) _Float16 Ks[128 * 32];   // [t][c]  (== B^T layout)
  __shared__ __align__(16) _Float16 Vt[32 * 128];   // [c][t]  (V transposed)
  __shared__ __align__(16) _Float16 Gs[128 * 32];   // [s][c]
  __shared__ __align__(16) _Float16 Ps[128 * 128];  // probs [s][t]

  const int tid = threadIdx.x;
  const int lane = tid & 31, wave = tid >> 5;

  // ---- stage 1a: TDM loads of Q, K, G (one issuing wave, DMA does the rest)
  const size_t base = (size_t)l * 1024 + h * 32;  // row s=0, in halves
  if (wave == 0) {
    tdm_load_2d_f16((unsigned int)(size_t)(void*)&Qs[0], qkvg + base + 0,
                    32u, 128u, (unsigned int)(L_DIM * 1024));
    tdm_load_2d_f16((unsigned int)(size_t)(void*)&Ks[0], qkvg + base + 256,
                    32u, 128u, (unsigned int)(L_DIM * 1024));
    tdm_load_2d_f16((unsigned int)(size_t)(void*)&Gs[0], qkvg + base + 768,
                    32u, 128u, (unsigned int)(L_DIM * 1024));
  }
  // ---- stage 1b: V loaded by all threads, stored transposed ----
  {
    const int s = tid >> 1;
    const int c0 = (tid & 1) * 16;
    const size_t rowbase = ((size_t)s * L_DIM + l) * 1024 + h * 32;
    v8h v0 = *(const v8h*)(qkvg + rowbase + 512 + c0);
    v8h v1 = *(const v8h*)(qkvg + rowbase + 512 + c0 + 8);
#pragma unroll
    for (int i = 0; i < 8; ++i) {
      Vt[(c0 + i) * 128 + s]     = v0[i];
      Vt[(c0 + i + 8) * 128 + s] = v1[i];
    }
  }
  if (wave == 0) {
    __builtin_amdgcn_s_wait_tensorcnt(0);
  }
  __syncthreads();

  // ---- stage 2: scores = Q K^T * scale, row softmax ----
  const float scale = 0.17677669529663687f;  // 1/sqrt(32)
  v16h aq = load_a_frag(Qs + wave * 16 * 32, 32);
  v8f sc[8];
#pragma unroll
  for (int j = 0; j < 8; ++j) {
    v16h bk = load_bT_frag(Ks + j * 16 * 32, 32);
    v8f z = {};
    sc[j] = wmma_f16(aq, bk, z);
  }

  const int nn = lane & 15, g = lane >> 4;
#pragma unroll
  for (int r = 0; r < 8; ++r) {
    float m = -1e30f;
#pragma unroll
    for (int j = 0; j < 8; ++j) {
      sc[j][r] *= scale;
      m = fmaxf(m, sc[j][r]);
    }
#pragma unroll
    for (int mask = 8; mask >= 1; mask >>= 1)
      m = fmaxf(m, __shfl_xor(m, mask, 32));
    float e[8];
    float sum = 0.f;
#pragma unroll
    for (int j = 0; j < 8; ++j) {
      e[j] = __expf(sc[j][r] - m);
      sum += e[j];
    }
#pragma unroll
    for (int mask = 8; mask >= 1; mask >>= 1)
      sum += __shfl_xor(sum, mask, 32);
    const float inv = 1.0f / sum;
    const int row = wave * 16 + r + 8 * g;
#pragma unroll
    for (int j = 0; j < 8; ++j)
      Ps[row * 128 + j * 16 + nn] = (_Float16)(e[j] * inv);
  }
  __syncthreads();

  // ---- stage 3: O = P * V (K = 128 = 4 x 32) ----
  v8f o0 = {}, o1 = {};
#pragma unroll
  for (int kb = 0; kb < 4; ++kb) {
    v16h ap = load_a_frag(Ps + wave * 16 * 128 + kb * 32, 128);
    v16h bv0 = load_bT_frag(Vt + 0 * 128 + kb * 32, 128);
    v16h bv1 = load_bT_frag(Vt + 16 * 128 + kb * 32, 128);
    o0 = wmma_f16(ap, bv0, o0);
    o1 = wmma_f16(ap, bv1, o1);
  }

  // ---- stage 4: gate with sigmoid(g + bg), write f16 ----
#pragma unroll
  for (int r = 0; r < 8; ++r) {
    const int s = wave * 16 + r + 8 * g;
    const int c0 = nn;
    const int c1 = 16 + nn;
    float g0 = (float)Gs[s * 32 + c0] + bg[h * 32 + c0];
    float g1 = (float)Gs[s * 32 + c1] + bg[h * 32 + c1];
    g0 = 1.0f / (1.0f + __expf(-g0));
    g1 = 1.0f / (1.0f + __expf(-g1));
    const size_t ob = ((size_t)s * L_DIM + l) * HC_DIM + h * 32;
    obuf[ob + c0] = (_Float16)(o0[r] * g0);
    obuf[ob + c1] = (_Float16)(o1[r] * g1);
  }
}

// ---------------------------------------------------------------------------
// Launcher
// ---------------------------------------------------------------------------
extern "C" void kernel_launch(void* const* d_in, const int* in_sizes, int n_in,
                              void* d_out, int out_size, void* d_ws,
                              size_t ws_size, hipStream_t stream) {
  const float* msa      = (const float*)d_in[0];
  const float* ln_scale = (const float*)d_in[1];
  const float* ln_bias  = (const float*)d_in[2];
  const float* wq       = (const float*)d_in[3];
  const float* wk       = (const float*)d_in[4];
  const float* wv       = (const float*)d_in[5];
  const float* wg       = (const float*)d_in[6];
  const float* bg       = (const float*)d_in[7];
  const float* wo       = (const float*)d_in[8];
  const float* bo       = (const float*)d_in[9];
  float* out = (float*)d_out;

  char* ws = (char*)d_ws;
  _Float16* x16  = (_Float16*)(ws);                        // 32 MB
  _Float16* Wf   = (_Float16*)(ws + 33554432);             // 512 KB
  _Float16* wo16 = (_Float16*)(ws + 33554432 + 524288);    // 128 KB
  _Float16* qkvg = (_Float16*)(ws + 34209792);             // 128 MB
  _Float16* obuf = (_Float16*)(ws + 34209792 + 134217728); // 32 MB

  // 1) LayerNorm -> f16
  ln_kernel<<<SL_DIM / 8, 256, 0, stream>>>(msa, ln_scale, ln_bias, x16);
  // 2) Pack weights -> f16
  pack_w_kernel<<<1280, 256, 0, stream>>>(wq, wk, wv, wg, wo, Wf, wo16);
  // 3) Fused QKVG projection: [65536,256] x [256,1024] -> f16
  gemm_kernel<1024, false><<<dim3(8, 1024), 256, 0, stream>>>(
      x16, Wf, qkvg, nullptr, nullptr);
  // 4) Attention per (head, residue), TDM-staged tiles
  attn_kernel<<<dim3(H_DIM, L_DIM), 256, 0, stream>>>(qkvg, bg, obuf);
  // 5) Output projection: [65536,256] x [256,256] + bo -> f32
  gemm_kernel<256, true><<<dim3(2, 1024), 256, 0, stream>>>(
      obuf, wo16, nullptr, out, bo);
}
